// GRKUCell_72584947302762
// MI455X (gfx1250) — matrix operand order repaired and checked by reading, not compiled, not run on hardware
//
#include <hip/hip_runtime.h>
#include <stdint.h>

// ---------------------------------------------------------------------------
// GRKU cell for MI455X (gfx1250, wave32).
//  - All GEMMs use v_wmma_f32_16x16x32_bf16 (bf16 in, f32 accum).
//  - Shared B tiles staged in LDS via async-to-LDS (ASYNCcnt) double-buffering.
//  - Unused reset gate (rg_*) skipped entirely.
//  - KAN spline einsums are fused GEMMs: A-fragments (cubic B-spline bases)
//    generated in registers, each lane packing two full 8-basis vectors.
// ---------------------------------------------------------------------------

#define B_ROWS 16384
#define IN_F   256
#define HDIM   512
#define MIDF   64
#define H2     1024
#define NBAS   8      // GRID_SIZE + ORDER
#define ORDER_K 3
#define GRID_LO_F (-1.0f)
#define GRID_H     0.4f
#define LN_EPS     1e-5f

#define BROW_U 40     // LDS row pitch in ushorts (80B: conflict-free frag reads)

typedef __attribute__((ext_vector_type(16))) __bf16 v16bf;
typedef __attribute__((ext_vector_type(8)))  float  v8f;

__device__ __forceinline__ unsigned short f2bf(float f) {
  union { float f; unsigned u; } a; a.f = f;
  unsigned r = a.u + 0x7FFFu + ((a.u >> 16) & 1u);   // round-to-nearest-even
  return (unsigned short)(r >> 16);
}
__device__ __forceinline__ float sigmoidf_(float x) { return 1.0f / (1.0f + __expf(-x)); }

// Cubic B-spline bases on the uniform reference grid (12 knots, h=0.4).
__device__ __forceinline__ void bases8(float x, float* bo) {
  float b[11];
#pragma unroll
  for (int j = 0; j < 11; ++j) {
    float t0 = GRID_LO_F + (float)(j - ORDER_K) * GRID_H;
    float t1 = t0 + GRID_H;
    b[j] = (x >= t0 && x < t1) ? 1.0f : 0.0f;
  }
#pragma unroll
  for (int k = 1; k <= ORDER_K; ++k) {
    const float inv = 1.0f / ((float)k * GRID_H);
#pragma unroll
    for (int j = 0; j < 11 - k; ++j) {
      float tj   = GRID_LO_F + (float)(j - ORDER_K) * GRID_H;
      float tjk1 = tj + (float)(k + 1) * GRID_H;
      b[j] = (x - tj) * inv * b[j] + (tjk1 - x) * inv * b[j + 1];
    }
  }
#pragma unroll
  for (int j = 0; j < 8; ++j) bo[j] = b[j];
}

// A-fragment (16x32 bf16): lane = M row; lanes 0-15 hold K {k0..k0+7, k0+16..23},
// lanes 16-31 hold K {k0+8..15, k0+24..31}.
__device__ __forceinline__ v16bf load_a_frag(const unsigned short* __restrict__ A,
                                             int lda, int m0, int k0, int lane) {
  int m  = m0 + (lane & 15);
  int kb = k0 + ((lane < 16) ? 0 : 8);
  const unsigned short* p = A + (size_t)m * lda + kb;
  union { uint4 u[2]; v16bf v; } u;
  u.u[0] = *(const uint4*)(p);
  u.u[1] = *(const uint4*)(p + 16);
  return u.v;
}

// --- async staging of the shared 64(N) x 32(K) bf16 B tile into LDS ---
// thread t of 128 covers a 32B half-row: row r = t/2, half hp = t%2.
// INST_OFFSET applies to both LDS and global addresses for async ops.
__device__ __forceinline__ void stage_b_tile(unsigned short* lds,
                                             const unsigned short* __restrict__ W,
                                             int ldw, int n0, int k0, int tid) {
  int r  = tid >> 1;
  int hp = tid & 1;
  const unsigned short* src = W + (size_t)(n0 + r) * ldw + k0 + hp * 16;
  unsigned dst = (unsigned)(uintptr_t)(lds + r * BROW_U + hp * 16);
  unsigned long long ga = (unsigned long long)(uintptr_t)src;
  asm volatile("global_load_async_to_lds_b128 %0, %1, off"
               :: "v"(dst), "v"(ga) : "memory");
  asm volatile("global_load_async_to_lds_b128 %0, %1, off offset:16"
               :: "v"(dst), "v"(ga) : "memory");
}

// B-fragment (32x16 bf16) read from the staged LDS tile: lane = N column;
// lanes 0-15 hold K {k0..k0+7, k0+16..23}, lanes 16-31 hold K {k0+8..15, k0+24..31}.
__device__ __forceinline__ v16bf lds_b_frag(const unsigned short* bt, int j, int lane) {
  const unsigned short* p = bt + (size_t)(16 * j + (lane & 15)) * BROW_U
                               + ((lane < 16) ? 0 : 8);
  union { uint4 u[2]; v16bf v; } u;
  u.u[0] = *(const uint4*)(p);
  u.u[1] = *(const uint4*)(p + 16);
  return u.v;
}

// ---------------- elementwise prep kernels ----------------
__global__ void k_f32_to_bf16(const float* __restrict__ s, unsigned short* __restrict__ d, int n) {
  int i = blockIdx.x * blockDim.x + threadIdx.x;
  if (i < n) d[i] = f2bf(s[i]);
}

__global__ void k_fold_spline(const float* __restrict__ sp, const float* __restrict__ sc,
                              unsigned short* __restrict__ d, int total) {
  int i = blockIdx.x * blockDim.x + threadIdx.x;
  if (i < total) d[i] = f2bf(sp[i] * sc[i >> 3]);
}

__global__ void k_combine(const float* __restrict__ xproj, const float* __restrict__ hprev,
                          float* __restrict__ cf, unsigned short* __restrict__ cb,
                          unsigned short* __restrict__ sb, int total) {
  int i = blockIdx.x * blockDim.x + threadIdx.x;
  if (i >= total) return;
  int b = i >> 10, c = i & 1023;
  float v = (c < HDIM) ? xproj[((size_t)b << 9) + c] : hprev[((size_t)b << 9) + (c - HDIM)];
  cf[i] = v;
  cb[i] = f2bf(v);
  sb[i] = f2bf(v * sigmoidf_(v));
}

__global__ void k_silu_to_bf16(const float* __restrict__ s, unsigned short* __restrict__ d, int n) {
  int i = blockIdx.x * blockDim.x + threadIdx.x;
  if (i < n) { float v = s[i]; d[i] = f2bf(v * sigmoidf_(v)); }
}

// ---------------- WMMA GEMM: O = A @ W^T + bias ----------------
// block = 128 (4 waves); wave tile 32(M) x 64(N); block tile 128 x 64.
// B tile (shared by all 4 waves) async-staged to LDS, double-buffered.
__global__ __launch_bounds__(128)
void k_gemm_bf16(const unsigned short* __restrict__ A, const unsigned short* __restrict__ W,
                 const float* __restrict__ bias, float* __restrict__ O,
                 int N, int K) {
  __shared__ unsigned short bt[2][64 * BROW_U];
  const int tid  = threadIdx.x;
  const int lane = tid & 31;
  const int wave = tid >> 5;
  const int m0 = blockIdx.x * 128 + wave * 32;
  const int n0 = blockIdx.y * 64;
  v8f c[2][4];
#pragma unroll
  for (int mi = 0; mi < 2; ++mi)
#pragma unroll
    for (int j = 0; j < 4; ++j) c[mi][j] = v8f{0, 0, 0, 0, 0, 0, 0, 0};

  stage_b_tile(bt[0], W, K, n0, 0, tid);
  int cur = 0;
  for (int k0 = 0; k0 < K; k0 += 32) {
    if (k0 + 32 < K) {
      stage_b_tile(bt[cur ^ 1], W, K, n0, k0 + 32, tid);
      asm volatile("s_wait_asynccnt 0x2" ::: "memory");
    } else {
      asm volatile("s_wait_asynccnt 0x0" ::: "memory");
    }
    __syncthreads();
    v16bf a0 = load_a_frag(A, K, m0,      k0, lane);
    v16bf a1 = load_a_frag(A, K, m0 + 16, k0, lane);
#pragma unroll
    for (int j = 0; j < 4; ++j) {
      v16bf b = lds_b_frag(bt[cur], j, lane);
      c[0][j] = __builtin_amdgcn_wmma_f32_16x16x32_bf16(false, a0, false, b, (short)0, c[0][j], false, false);
      c[1][j] = __builtin_amdgcn_wmma_f32_16x16x32_bf16(false, a1, false, b, (short)0, c[1][j], false, false);
    }
    __syncthreads();
    cur ^= 1;
  }

  const int nl = lane & 15;
#pragma unroll
  for (int mi = 0; mi < 2; ++mi) {
    const int mb = m0 + 16 * mi + ((lane < 16) ? 0 : 8);
#pragma unroll
    for (int j = 0; j < 4; ++j) {
      int n = n0 + 16 * j + nl;
      float bv = bias[n];
#pragma unroll
      for (int r = 0; r < 8; ++r) O[(size_t)(mb + r) * N + n] = c[mi][j][r] + bv;
    }
  }
}

// ---------------- fused KAN layer: O = silu(x)@BW^T + bases(x)@SW^T ----------------
// Same LDS-staged B pipeline; spline A-fragments built in registers
// (each lane evaluates 2 full 8-basis vectors per M-fragment per K-step).
template<int INF_T, int NOUT_T>
__global__ __launch_bounds__(128)
void k_kan(const float* __restrict__ Xf, const unsigned short* __restrict__ Xsilu,
           const unsigned short* __restrict__ BW, const unsigned short* __restrict__ SW,
           float* __restrict__ O) {
  __shared__ unsigned short bt[2][64 * BROW_U];
  const int tid  = threadIdx.x;
  const int lane = tid & 31;
  const int wave = tid >> 5;
  const int m0 = blockIdx.x * 128 + wave * 32;
  const int n0 = blockIdx.y * 64;
  v8f c[2][4];
#pragma unroll
  for (int mi = 0; mi < 2; ++mi)
#pragma unroll
    for (int j = 0; j < 4; ++j) c[mi][j] = v8f{0, 0, 0, 0, 0, 0, 0, 0};

  // ---- base term, K = INF_T ----
  stage_b_tile(bt[0], BW, INF_T, n0, 0, tid);
  int cur = 0;
  for (int k0 = 0; k0 < INF_T; k0 += 32) {
    if (k0 + 32 < INF_T) {
      stage_b_tile(bt[cur ^ 1], BW, INF_T, n0, k0 + 32, tid);
      asm volatile("s_wait_asynccnt 0x2" ::: "memory");
    } else {
      asm volatile("s_wait_asynccnt 0x0" ::: "memory");
    }
    __syncthreads();
    v16bf a0 = load_a_frag(Xsilu, INF_T, m0,      k0, lane);
    v16bf a1 = load_a_frag(Xsilu, INF_T, m0 + 16, k0, lane);
#pragma unroll
    for (int j = 0; j < 4; ++j) {
      v16bf b = lds_b_frag(bt[cur], j, lane);
      c[0][j] = __builtin_amdgcn_wmma_f32_16x16x32_bf16(false, a0, false, b, (short)0, c[0][j], false, false);
      c[1][j] = __builtin_amdgcn_wmma_f32_16x16x32_bf16(false, a1, false, b, (short)0, c[1][j], false, false);
    }
    __syncthreads();
    cur ^= 1;
  }

  // ---- spline term, K = INF_T * 8 ; feature f = k/8 ----
  const float* xr0 = Xf + (size_t)(m0 +      (lane & 15)) * INF_T;
  const float* xr1 = Xf + (size_t)(m0 + 16 + (lane & 15)) * INF_T;
  stage_b_tile(bt[0], SW, INF_T * NBAS, n0, 0, tid);
  cur = 0;
  for (int k0 = 0; k0 < INF_T * NBAS; k0 += 32) {
    if (k0 + 32 < INF_T * NBAS) {
      stage_b_tile(bt[cur ^ 1], SW, INF_T * NBAS, n0, k0 + 32, tid);
      asm volatile("s_wait_asynccnt 0x2" ::: "memory");
    } else {
      asm volatile("s_wait_asynccnt 0x0" ::: "memory");
    }
    __syncthreads();
    int f0 = (k0 >> 3) + ((lane < 16) ? 0 : 1);
    union { unsigned short us[16]; v16bf v; } u0, u1;
    {
      float ba[8], bb[8];
      bases8(xr0[f0], ba);
      bases8(xr0[f0 + 2], bb);
#pragma unroll
      for (int j = 0; j < 8; ++j) { u0.us[j] = f2bf(ba[j]); u0.us[8 + j] = f2bf(bb[j]); }
      bases8(xr1[f0], ba);
      bases8(xr1[f0 + 2], bb);
#pragma unroll
      for (int j = 0; j < 8; ++j) { u1.us[j] = f2bf(ba[j]); u1.us[8 + j] = f2bf(bb[j]); }
    }
#pragma unroll
    for (int j = 0; j < 4; ++j) {
      v16bf b = lds_b_frag(bt[cur], j, lane);
      c[0][j] = __builtin_amdgcn_wmma_f32_16x16x32_bf16(false, u0.v, false, b, (short)0, c[0][j], false, false);
      c[1][j] = __builtin_amdgcn_wmma_f32_16x16x32_bf16(false, u1.v, false, b, (short)0, c[1][j], false, false);
    }
    __syncthreads();
    cur ^= 1;
  }

  const int nl = lane & 15;
#pragma unroll
  for (int mi = 0; mi < 2; ++mi) {
    const int mb = m0 + 16 * mi + ((lane < 16) ? 0 : 8);
#pragma unroll
    for (int j = 0; j < 4; ++j) {
      int n = n0 + 16 * j + nl;
#pragma unroll
      for (int r = 0; r < 8; ++r) O[(size_t)(mb + r) * NOUT_T + n] = c[mi][j][r];
    }
  }
}

// ---------------- row LayerNorm + sigmoid (update gate) ----------------
__global__ __launch_bounds__(256)
void k_ln_sigmoid(const float* __restrict__ in, const float* __restrict__ g,
                  const float* __restrict__ be, float* __restrict__ out) {
  const int row = blockIdx.x, t = threadIdx.x;
  const size_t base = (size_t)row * HDIM;
  float v0 = in[base + t], v1 = in[base + t + 256];
  __shared__ float s1[256], s2[256];
  s1[t] = v0 + v1; s2[t] = v0 * v0 + v1 * v1;
  __syncthreads();
  for (int off = 128; off > 0; off >>= 1) {
    if (t < off) { s1[t] += s1[t + off]; s2[t] += s2[t + off]; }
    __syncthreads();
  }
  float mu  = s1[0] * (1.0f / HDIM);
  float var = s2[0] * (1.0f / HDIM) - mu * mu;
  float rs  = rsqrtf(var + LN_EPS);
  out[base + t]       = sigmoidf_((v0 - mu) * rs * g[t] + be[t]);
  out[base + t + 256] = sigmoidf_((v1 - mu) * rs * g[t + 256] + be[t + 256]);
}

// ---------------- final: (kan2+resid) -> LN -> LN -> silu -> gate ----------------
__global__ __launch_bounds__(256)
void k_final(const float* __restrict__ k2o, const float* __restrict__ resid,
             const float* __restrict__ g, const float* __restrict__ be,
             const float* __restrict__ upd, const float* __restrict__ hp,
             float* __restrict__ out) {
  const int row = blockIdx.x, t = threadIdx.x;
  const size_t base = (size_t)row * HDIM;
  float v0 = k2o[base + t] + resid[base + t];
  float v1 = k2o[base + t + 256] + resid[base + t + 256];
  __shared__ float s1[256], s2[256];

  s1[t] = v0 + v1; s2[t] = v0 * v0 + v1 * v1;
  __syncthreads();
  for (int off = 128; off > 0; off >>= 1) {
    if (t < off) { s1[t] += s1[t + off]; s2[t] += s2[t + off]; }
    __syncthreads();
  }
  float mu  = s1[0] * (1.0f / HDIM);
  float var = s2[0] * (1.0f / HDIM) - mu * mu;
  float rs  = rsqrtf(var + LN_EPS);
  float y0 = (v0 - mu) * rs * g[t] + be[t];
  float y1 = (v1 - mu) * rs * g[t + 256] + be[t + 256];
  __syncthreads();

  s1[t] = y0 + y1; s2[t] = y0 * y0 + y1 * y1;
  __syncthreads();
  for (int off = 128; off > 0; off >>= 1) {
    if (t < off) { s1[t] += s1[t + off]; s2[t] += s2[t + off]; }
    __syncthreads();
  }
  mu  = s1[0] * (1.0f / HDIM);
  var = s2[0] * (1.0f / HDIM) - mu * mu;
  rs  = rsqrtf(var + LN_EPS);
  float z0 = (y0 - mu) * rs * g[t] + be[t];
  float z1 = (y1 - mu) * rs * g[t + 256] + be[t + 256];
  z0 = z0 * sigmoidf_(z0);
  z1 = z1 * sigmoidf_(z1);
  float u0 = upd[base + t], u1 = upd[base + t + 256];
  out[base + t]       = (1.0f - u0) * hp[base + t]       + u0 * z0;
  out[base + t + 256] = (1.0f - u1) * hp[base + t + 256] + u1 * z1;
}

// ---------------------------------------------------------------------------
extern "C" void kernel_launch(void* const* d_in, const int* in_sizes, int n_in,
                              void* d_out, int out_size, void* d_ws, size_t ws_size,
                              hipStream_t stream) {
  (void)in_sizes; (void)n_in; (void)out_size; (void)ws_size;
  const float* x        = (const float*)d_in[0];
  const float* h_prev   = (const float*)d_in[1];
  const float* ip_w     = (const float*)d_in[2];
  const float* ip_b     = (const float*)d_in[3];
  // d_in[4..7] = reset gate params: unused by the reference output -> skipped
  const float* ug_w     = (const float*)d_in[8];
  const float* ug_b     = (const float*)d_in[9];
  const float* ug_g     = (const float*)d_in[10];
  const float* ug_beta  = (const float*)d_in[11];
  const float* k1_base  = (const float*)d_in[12];
  const float* k1_spline= (const float*)d_in[13];
  const float* k1_scaler= (const float*)d_in[14];
  const float* k2_base  = (const float*)d_in[16];
  const float* k2_spline= (const float*)d_in[17];
  const float* k2_scaler= (const float*)d_in[18];
  const float* cn_g     = (const float*)d_in[20];
  const float* cn_b     = (const float*)d_in[21];
  const float* rp_w     = (const float*)d_in[22];
  const float* rp_b     = (const float*)d_in[23];
  float* out = (float*)d_out;

  char* p = (char*)d_ws;
  auto alloc = [&](size_t bytes) -> char* {
    char* r = p; p += (bytes + 255) & ~(size_t)255; return r;
  };
  unsigned short* x_bf   = (unsigned short*)alloc((size_t)B_ROWS * IN_F * 2);
  unsigned short* ipw_bf = (unsigned short*)alloc((size_t)HDIM * IN_F * 2);
  unsigned short* ugw_bf = (unsigned short*)alloc((size_t)HDIM * H2 * 2);
  unsigned short* rpw_bf = (unsigned short*)alloc((size_t)HDIM * H2 * 2);
  unsigned short* k1b_bf = (unsigned short*)alloc((size_t)MIDF * H2 * 2);
  unsigned short* w1s_bf = (unsigned short*)alloc((size_t)MIDF * H2 * NBAS * 2);
  unsigned short* k2b_bf = (unsigned short*)alloc((size_t)HDIM * MIDF * 2);
  unsigned short* w2s_bf = (unsigned short*)alloc((size_t)HDIM * MIDF * NBAS * 2);
  float*          xproj  = (float*)alloc((size_t)B_ROWS * HDIM * 4);
  float*          comb_f = (float*)alloc((size_t)B_ROWS * H2 * 4);
  unsigned short* comb_bf= (unsigned short*)alloc((size_t)B_ROWS * H2 * 2);
  unsigned short* silu_bf= (unsigned short*)alloc((size_t)B_ROWS * H2 * 2);
  float*          ug_raw = (float*)alloc((size_t)B_ROWS * HDIM * 4);
  float*          update = (float*)alloc((size_t)B_ROWS * HDIM * 4);
  float*          resid  = (float*)alloc((size_t)B_ROWS * HDIM * 4);
  float*          mid    = (float*)alloc((size_t)B_ROWS * MIDF * 4);
  unsigned short* mid_sbf= (unsigned short*)alloc((size_t)B_ROWS * MIDF * 2);
  float*          k2out  = (float*)alloc((size_t)B_ROWS * HDIM * 4);

  auto cvt = [&](const float* s, unsigned short* d, int n) {
    k_f32_to_bf16<<<(n + 255) / 256, 256, 0, stream>>>(s, d, n);
  };

  // --- weight / input prep ---
  cvt(x, x_bf, B_ROWS * IN_F);
  cvt(ip_w, ipw_bf, HDIM * IN_F);
  cvt(ug_w, ugw_bf, HDIM * H2);
  cvt(rp_w, rpw_bf, HDIM * H2);
  cvt(k1_base, k1b_bf, MIDF * H2);
  cvt(k2_base, k2b_bf, HDIM * MIDF);
  {
    int t1 = MIDF * H2 * NBAS;
    k_fold_spline<<<(t1 + 255) / 256, 256, 0, stream>>>(k1_spline, k1_scaler, w1s_bf, t1);
    int t2 = HDIM * MIDF * NBAS;
    k_fold_spline<<<(t2 + 255) / 256, 256, 0, stream>>>(k2_spline, k2_scaler, w2s_bf, t2);
  }

  // --- x_proj = x @ ip_w^T + ip_b ---
  k_gemm_bf16<<<dim3(B_ROWS / 128, HDIM / 64), 128, 0, stream>>>(x_bf, ipw_bf, ip_b, xproj, HDIM, IN_F);

  // --- combined = [x_proj, h_prev] (f32 + bf16 + silu-bf16) ---
  {
    int total = B_ROWS * H2;
    k_combine<<<(total + 255) / 256, 256, 0, stream>>>(xproj, h_prev, comb_f, comb_bf, silu_bf, total);
  }

  // --- update gate GEMM + residual GEMM ---
  k_gemm_bf16<<<dim3(B_ROWS / 128, HDIM / 64), 128, 0, stream>>>(comb_bf, ugw_bf, ug_b, ug_raw, HDIM, H2);
  k_gemm_bf16<<<dim3(B_ROWS / 128, HDIM / 64), 128, 0, stream>>>(comb_bf, rpw_bf, rp_b, resid, HDIM, H2);

  // --- update = sigmoid(LN(ug_raw)) ---
  k_ln_sigmoid<<<B_ROWS, 256, 0, stream>>>(ug_raw, ug_g, ug_beta, update);

  // --- KAN1: combined (H2) -> mid (MIDF) ---
  k_kan<H2, MIDF><<<dim3(B_ROWS / 128, MIDF / 64), 128, 0, stream>>>(comb_f, silu_bf, k1b_bf, w1s_bf, mid);

  // --- mid silu in bf16 ---
  {
    int n = B_ROWS * MIDF;
    k_silu_to_bf16<<<(n + 255) / 256, 256, 0, stream>>>(mid, mid_sbf, n);
  }

  // --- KAN2: mid (MIDF) -> h_cand raw (HDIM) ---
  k_kan<MIDF, HDIM><<<dim3(B_ROWS / 128, HDIM / 64), 128, 0, stream>>>(mid, mid_sbf, k2b_bf, w2s_bf, k2out);

  // --- final: +resid, LN, LN, silu, gate ---
  k_final<<<B_ROWS, 256, 0, stream>>>(k2out, resid, cn_g, cn_b, update, h_prev, out);
}
